// DRAW_35459249996421
// MI455X (gfx1250) — matrix operand order repaired and checked by reading, not compiled, not run on hardware
//
#include <hip/hip_runtime.h>

// ---------------------------------------------------------------------------
// DRAW (B=512, H=W=64, T=32, L=1024, Z=128, RS=WS=12) for gfx1250 / MI455X.
// Heavy GEMMs: v_wmma_f32_16x16x32_bf16, A-tile staged to LDS via TDM
// (tensor_load_to_lds + s_wait_tensorcnt), triple-buffered B stream.
// ---------------------------------------------------------------------------

#define B_   512
#define HW_  4096      // H*W
#define L_   1024
#define Z_   128
#define T_   32
#define NF_  12        // RS == WS == 12
#define KENC_ 2336     // 288 (r) + 1024 (h_dec) + 1024 (h_enc)
#define KDEC_ 1152     // 128 (z) + 1024 (h_dec)

typedef __attribute__((ext_vector_type(16))) __bf16 v16bf;
typedef __attribute__((ext_vector_type(8)))  float  v8f;
typedef __attribute__((ext_vector_type(4)))  float  f32x4;
typedef __attribute__((ext_vector_type(4)))  unsigned int u32x4;
typedef __attribute__((ext_vector_type(8)))  int    i32x8;
typedef __attribute__((ext_vector_type(4)))  int    i32x4;

__device__ __forceinline__ float sigmoidf_(float x) { return 1.0f / (1.0f + __expf(-x)); }

// ------------------------- WMMA GEMM with TDM A-staging --------------------
// C(M,N) = A(M,K) @ W(N,K)^T.  A, W bf16 row-major; C f32 row-major.
// K % 32 == 0, M % 32 == 0, N % 256 == 0.
// Block = 256 threads = 8 waves arranged 2(M) x 4(N); wave tile 16x64.
// A tile (32 x K) staged into dynamic LDS by the Tensor Data Mover.

#define LOAD_SLAB(AF, BF, KOFF)                                         \
  {                                                                     \
    const int kk_ = (KOFF);                                             \
    AF.f[0] = *(const f32x4*)(arow + kk_ + half * 8);                   \
    AF.f[1] = *(const f32x4*)(arow + kk_ + 16 + half * 8);              \
    _Pragma("unroll")                                                   \
    for (int j = 0; j < 4; ++j) {                                       \
      BF[j].f[0] = *(const f32x4*)(brow[j] + kk_ + half * 8);           \
      BF[j].f[1] = *(const f32x4*)(brow[j] + kk_ + 16 + half * 8);      \
    }                                                                   \
  }

#if defined(__gfx1250__)
#define WMMA_SLAB(AF, BF)                                               \
  _Pragma("unroll")                                                     \
  for (int j = 0; j < 4; ++j)                                           \
    acc[j] = __builtin_amdgcn_wmma_f32_16x16x32_bf16(                   \
        false, AF.v, false, BF[j].v, (short)0, acc[j], false, false);
#else
#define WMMA_SLAB(AF, BF) ((void)0);
#endif

__global__ __launch_bounds__(256) void k_gemm_bf16(
    const __bf16* __restrict__ A, const __bf16* __restrict__ Wt,
    float* __restrict__ C, int M, int N, int K) {
  extern __shared__ __bf16 As[];             // 32 * K bf16
  const int wave = threadIdx.x >> 5;
  const int lane = threadIdx.x & 31;
  const int wm = wave >> 2, wn = wave & 3;
  const int mb = blockIdx.x * 32;
  const int m0 = mb + wm * 16;
  const int n0 = blockIdx.y * 256 + wn * 64;
  const int half = lane >> 4;                // 0 | 1
  const int l = lane & 15;

  // ---- stage A tile (32 rows x K cols) into LDS
  const __bf16* Atile = A + (size_t)mb * K;
#if defined(__gfx1250__) && __has_builtin(__builtin_amdgcn_tensor_load_to_lds)
  if (threadIdx.x == 0) {
    unsigned lds_off = (unsigned)(size_t)(void*)As;
    unsigned long long ga = (unsigned long long)(size_t)Atile;
    // D# group 0: count=1 | lds_addr | global_addr | type=2 ("image")
    u32x4 g0;
    g0[0] = 1u;
    g0[1] = lds_off;
    g0[2] = (unsigned)(ga & 0xffffffffu);
    g0[3] = (unsigned)((ga >> 32) & 0x01ffffffu) | (2u << 30);
    // D# group 1: data_size=1 (2B); tensor_dim0=K; tensor_dim1=32;
    //             tile_dim0=K; tile_dim1=32; tensor_dim0_stride=K
    unsigned uK = (unsigned)K;
    i32x8 g1;
    g1[0] = (int)(1u << 16);                              // data_size = 2 bytes
    g1[1] = (int)((uK & 0xffffu) << 16);                  // tensor_dim0[15:0]
    g1[2] = (int)(((uK >> 16) & 0xffffu) | (32u << 16));  // dim0[31:16] | tensor_dim1[15:0]
    g1[3] = (int)((uK & 0xffffu) << 16);                  // dim1[31:16]=0 | tile_dim0
    g1[4] = 32;                                           // tile_dim1=32, tile_dim2=0
    g1[5] = (int)uK;                                      // tensor_dim0_stride[31:0]
    g1[6] = 0;                                            // stride0[47:32] | stride1 lo
    g1[7] = 0;
    i32x4 gz = {0, 0, 0, 0};
#if __clang_major__ >= 23
    i32x8 gz8 = {0, 0, 0, 0, 0, 0, 0, 0};
    __builtin_amdgcn_tensor_load_to_lds(g0, g1, gz, gz, gz8, 0);
#else
    __builtin_amdgcn_tensor_load_to_lds(g0, g1, gz, gz, 0);
#endif
    __builtin_amdgcn_s_wait_tensorcnt(0);
  }
  __syncthreads();
#else
  for (int i = threadIdx.x; i < (32 * K) >> 3; i += 256)
    ((f32x4*)As)[i] = ((const f32x4*)Atile)[i];
  __syncthreads();
#endif

  v8f vzero = {0.f, 0.f, 0.f, 0.f, 0.f, 0.f, 0.f, 0.f};
  v8f acc[4];
#pragma unroll
  for (int j = 0; j < 4; ++j) acc[j] = vzero;

  // A fragments come from LDS (ds_load_b128); B streams from L2
  const __bf16* arow = As + (size_t)(wm * 16 + l) * K;
  const __bf16* brow[4];
#pragma unroll
  for (int j = 0; j < 4; ++j) brow[j] = Wt + (size_t)(n0 + j * 16 + l) * K;

  union Frag { v16bf v; f32x4 f[2]; };

  // ---- triple-buffered K loop: load target regs are never live at the
  //      point the scheduler wants to hoist the loads -> no rotation movs.
  const int nsl = K >> 5;                    // K/32 slabs; nsl >= 32 here
  Frag a0, b0[4], a1, b1[4], a2, b2[4];
  LOAD_SLAB(a0, b0, 0);
  LOAD_SLAB(a1, b1, 32);
  int s = 0;
  for (; s + 4 < nsl; s += 3) {
    LOAD_SLAB(a2, b2, (s + 2) << 5);
    WMMA_SLAB(a0, b0);                       // slab s
    LOAD_SLAB(a0, b0, (s + 3) << 5);
    WMMA_SLAB(a1, b1);                       // slab s+1
    if (s + 5 < nsl) {
#pragma unroll
      for (int j = 0; j < 4; ++j) __builtin_prefetch(brow[j] + ((s + 5) << 5), 0, 3);
    }
    LOAD_SLAB(a1, b1, (s + 4) << 5);
    WMMA_SLAB(a2, b2);                       // slab s+2
  }
  {
    const int rem = nsl - s;                 // 2, 3 or 4
    if (rem == 2) {
      WMMA_SLAB(a0, b0);
      WMMA_SLAB(a1, b1);
    } else if (rem == 3) {
      LOAD_SLAB(a2, b2, (s + 2) << 5);
      WMMA_SLAB(a0, b0);
      WMMA_SLAB(a1, b1);
      WMMA_SLAB(a2, b2);
    } else {
      LOAD_SLAB(a2, b2, (s + 2) << 5);
      WMMA_SLAB(a0, b0);
      LOAD_SLAB(a0, b0, (s + 3) << 5);
      WMMA_SLAB(a1, b1);
      WMMA_SLAB(a2, b2);
      WMMA_SLAB(a0, b0);
    }
  }

  // C/D layout: lanes 0-15 rows m0..m0+7 (elem e), lanes 16-31 rows m0+8..m0+15
#pragma unroll
  for (int j = 0; j < 4; ++j) {
    int col = n0 + j * 16 + l;
#pragma unroll
    for (int e = 0; e < 8; ++e) {
      int row = m0 + half * 8 + e;
      C[(size_t)row * N + col] = acc[j][e];
    }
  }
}

// ------------------------- weight conversion -------------------------------
__global__ void k_cvt(const float* __restrict__ s, __bf16* __restrict__ d, size_t n) {
  size_t i = (size_t)blockIdx.x * blockDim.x + threadIdx.x;
  if (i < n) d[i] = (__bf16)s[i];
}

// dst row n = [w0 row n (k0 cols) | w1 row n (k1 cols)]
__global__ void k_cvt2(const float* __restrict__ w0, int k0,
                       const float* __restrict__ w1, int k1,
                       __bf16* __restrict__ dst, int nrows) {
  const int K = k0 + k1;
  size_t i = (size_t)blockIdx.x * blockDim.x + threadIdx.x;
  if (i >= (size_t)nrows * K) return;
  int n = (int)(i / K), k = (int)(i % K);
  float v = (k < k0) ? w0[(size_t)n * k0 + k] : w1[(size_t)n * k1 + (k - k0)];
  dst[i] = (__bf16)v;
}

__global__ void k_zero(float* p, size_t n) {
  size_t i = (size_t)blockIdx.x * blockDim.x + threadIdx.x;
  if (i < n) p[i] = 0.f;
}

// ------------------------- pointwise ---------------------------------------
__global__ void k_xhat(const float* __restrict__ x, const float* __restrict__ c,
                       float* __restrict__ xhat) {
  size_t i = (size_t)blockIdx.x * blockDim.x + threadIdx.x;
  if (i < (size_t)B_ * HW_) xhat[i] = x[i] - sigmoidf_(c[i]);
}

// out[b][n] = bias[n] + sum_k h[b][k] * W[n][k]   (small N)
__global__ void k_linear(const float* __restrict__ h, const float* __restrict__ W,
                         const float* __restrict__ bias, float* __restrict__ out,
                         int N, int K) {
  int i = blockIdx.x * blockDim.x + threadIdx.x;
  if (i >= B_ * N) return;
  int b = i / N, n = i % N;
  const float* hr = h + (size_t)b * K;
  const float* wr = W + (size_t)n * K;
  float s = bias[n];
  for (int k = 0; k < K; ++k) s += hr[k] * wr[k];
  out[i] = s;
}

// one block per (b, n) filter row; 64 threads (W == H == 64)
__global__ __launch_bounds__(64) void k_filterbank(
    const float* __restrict__ a, float* __restrict__ Fx, float* __restrict__ Fy,
    float* __restrict__ gamma, float gsign) {
  int b = blockIdx.x / NF_;
  int n = blockIdx.x % NF_;
  float gx = a[b * 5 + 0], gy = a[b * 5 + 1];
  float lv = a[b * 5 + 2], ld = a[b * 5 + 3], lg = a[b * 5 + 4];
  gx = 0.5f * 65.0f * (gx + 1.0f);
  gy = 0.5f * 65.0f * (gy + 1.0f);
  float delta = (63.0f / 11.0f) * __expf(ld);
  float off = (float)(n + 1) - 6.5f;     // arange(1..12) - 0.5*13
  float mux = gx + off * delta;
  float muy = gy + off * delta;
  float inv2var = 0.5f * __expf(-lv);
  int w = threadIdx.x;                   // 0..63
  float dx = (float)(w + 1) - mux;
  float dy = (float)(w + 1) - muy;
  float fx = __expf(-inv2var * dx * dx);
  float fy = __expf(-inv2var * dy * dy);
  __shared__ float sx[64], sy[64];
  sx[w] = fx; sy[w] = fy;
  __syncthreads();
  for (int s = 32; s > 0; s >>= 1) {
    if (w < s) { sx[w] += sx[w + s]; sy[w] += sy[w + s]; }
    __syncthreads();
  }
  float sumx = sx[0] + 64e-8f;           // sum(Fx + 1e-8) over W=64
  float sumy = sy[0] + 64e-8f;
  Fx[((size_t)b * NF_ + n) * 64 + w] = fx / sumx;
  Fy[((size_t)b * NF_ + n) * 64 + w] = fy / sumy;
  if (n == 0 && w == 0) gamma[b] = __expf(gsign * lg);
}

// r[b] = gamma * concat(Fy @ X @ Fx^T, Fy @ Xhat @ Fx^T)  -> (B, 288)
__global__ __launch_bounds__(256) void k_read(
    const float* __restrict__ x, const float* __restrict__ xhat,
    const float* __restrict__ Fx, const float* __restrict__ Fy,
    const float* __restrict__ gamma, float* __restrict__ r) {
  int b = blockIdx.x;
  __shared__ float tmp[NF_ * 64];
  const float* fx = Fx + (size_t)b * NF_ * 64;
  const float* fy = Fy + (size_t)b * NF_ * 64;
  float g = gamma[b];
  for (int img = 0; img < 2; ++img) {
    const float* X = (img == 0 ? x : xhat) + (size_t)b * HW_;
    for (int idx = threadIdx.x; idx < NF_ * 64; idx += 256) {
      int n = idx >> 6, w = idx & 63;
      float s = 0.f;
      for (int h = 0; h < 64; ++h) s += fy[n * 64 + h] * X[h * 64 + w];
      tmp[idx] = s;
    }
    __syncthreads();
    for (int idx = threadIdx.x; idx < NF_ * NF_; idx += 256) {
      int n = idx / NF_, m = idx % NF_;
      float s = 0.f;
      for (int w = 0; w < 64; ++w) s += tmp[n * 64 + w] * fx[m * 64 + w];
      r[(size_t)b * (2 * NF_ * NF_) + img * (NF_ * NF_) + idx] = g * s;
    }
    __syncthreads();
  }
}

// c[b] += exp(-loggamma) * (Fy^T @ w @ Fx)
__global__ __launch_bounds__(256) void k_write(
    const float* __restrict__ wv, const float* __restrict__ Fx,
    const float* __restrict__ Fy, const float* __restrict__ ginv,
    float* __restrict__ c) {
  int b = blockIdx.x;
  __shared__ float tmp[64 * NF_];
  const float* fx = Fx + (size_t)b * NF_ * 64;
  const float* fy = Fy + (size_t)b * NF_ * 64;
  const float* w = wv + (size_t)b * (NF_ * NF_);
  float g = ginv[b];
  for (int idx = threadIdx.x; idx < 64 * NF_; idx += 256) {
    int h = idx / NF_, m = idx % NF_;
    float s = 0.f;
    for (int n = 0; n < NF_; ++n) s += fy[n * 64 + h] * w[n * NF_ + m];
    tmp[idx] = s;
  }
  __syncthreads();
  for (int idx = threadIdx.x; idx < HW_; idx += 256) {
    int h = idx >> 6, wc = idx & 63;
    float s = 0.f;
    for (int m = 0; m < NF_; ++m) s += tmp[h * NF_ + m] * fx[m * 64 + wc];
    c[(size_t)b * HW_ + idx] += g * s;
  }
}

// gates G(B,4096) -> update h, c (B,1024)
__global__ void k_lstm(const float* __restrict__ G, const float* __restrict__ bih,
                       const float* __restrict__ bhh, float* __restrict__ h,
                       float* __restrict__ c) {
  int i = blockIdx.x * blockDim.x + threadIdx.x;
  if (i >= B_ * L_) return;
  int b = i >> 10, n = i & 1023;
  const float* g = G + (size_t)b * (4 * L_);
  float gi = g[n]          + bih[n]          + bhh[n];
  float gf = g[L_ + n]     + bih[L_ + n]     + bhh[L_ + n];
  float gg = g[2 * L_ + n] + bih[2 * L_ + n] + bhh[2 * L_ + n];
  float go = g[3 * L_ + n] + bih[3 * L_ + n] + bhh[3 * L_ + n];
  float cf = sigmoidf_(gf) * c[i] + sigmoidf_(gi) * tanhf(gg);
  c[i] = cf;
  h[i] = sigmoidf_(go) * tanhf(cf);
}

// pack [r(288) | h_dec(1024) | h_enc(1024)] -> bf16 (B, 2336)
__global__ void k_pack_enc(const float* __restrict__ r, const float* __restrict__ hdec,
                           const float* __restrict__ henc, __bf16* __restrict__ A) {
  size_t i = (size_t)blockIdx.x * blockDim.x + threadIdx.x;
  if (i >= (size_t)B_ * KENC_) return;
  int b = (int)(i / KENC_), k = (int)(i % KENC_);
  float v;
  if (k < 288)       v = r[(size_t)b * 288 + k];
  else if (k < 1312) v = hdec[(size_t)b * L_ + (k - 288)];
  else               v = henc[(size_t)b * L_ + (k - 1312)];
  A[i] = (__bf16)v;
}

// pack [z(128) | h_dec(1024)] -> bf16 (B, 1152)
__global__ void k_pack_dec(const float* __restrict__ z, const float* __restrict__ hdec,
                           __bf16* __restrict__ A) {
  size_t i = (size_t)blockIdx.x * blockDim.x + threadIdx.x;
  if (i >= (size_t)B_ * KDEC_) return;
  int b = (int)(i / KDEC_), k = (int)(i % KDEC_);
  float v = (k < Z_) ? z[(size_t)b * Z_ + k] : hdec[(size_t)b * L_ + (k - Z_)];
  A[i] = (__bf16)v;
}

__device__ __forceinline__ unsigned pcg_(unsigned v) {
  v = v * 747796405u + 2891336453u;
  v = ((v >> ((v >> 28) + 4u)) ^ v) * 277803737u;
  return (v >> 22) ^ v;
}

// z = mu + sig * eps; also emit mu/sig into d_out at (B, Z, T) layout slots
__global__ void k_z(const float* __restrict__ stats, const float* __restrict__ zb,
                    float* __restrict__ z, float* __restrict__ out, int t) {
  int i = blockIdx.x * blockDim.x + threadIdx.x;
  if (i >= B_ * Z_) return;
  float mu = stats[(size_t)(i >> 7) * (2 * Z_) + (i & 127)] + zb[i & 127];
  float ls = stats[(size_t)(i >> 7) * (2 * Z_) + Z_ + (i & 127)] + zb[Z_ + (i & 127)];
  float sig = __expf(ls);
  unsigned s = (unsigned)(i * T_ + t);
  unsigned u1 = pcg_(2u * s + 1u);
  unsigned u2 = pcg_(2u * s + 0x9E3779B9u);
  float f1 = ((float)u1 + 1.0f) * 2.3283064e-10f;
  float f2 = (float)u2 * 2.3283064e-10f;
  float eps = __fsqrt_rn(-2.0f * __logf(f1)) * __cosf(6.2831853f * f2);
  z[i] = mu + sig * eps;
  const size_t C_SZ = (size_t)B_ * HW_;
  const size_t ZT = (size_t)B_ * Z_ * T_;
  out[C_SZ + (size_t)i * T_ + t] = mu;
  out[C_SZ + ZT + (size_t)i * T_ + t] = sig;
}

// ---------------------------------------------------------------------------
extern "C" void kernel_launch(void* const* d_in, const int* in_sizes, int n_in,
                              void* d_out, int out_size, void* d_ws, size_t ws_size,
                              hipStream_t stream) {
  const float* x       = (const float*)d_in[0];
  const float* encWih  = (const float*)d_in[1];
  const float* encWhh  = (const float*)d_in[2];
  const float* enc_bih = (const float*)d_in[3];
  const float* enc_bhh = (const float*)d_in[4];
  const float* decWih  = (const float*)d_in[5];
  const float* decWhh  = (const float*)d_in[6];
  const float* dec_bih = (const float*)d_in[7];
  const float* dec_bhh = (const float*)d_in[8];
  const float* zW      = (const float*)d_in[9];
  const float* zb      = (const float*)d_in[10];
  const float* wW      = (const float*)d_in[11];
  const float* wb      = (const float*)d_in[12];
  const float* raW     = (const float*)d_in[13];
  const float* rab     = (const float*)d_in[14];
  const float* waW     = (const float*)d_in[15];
  const float* wab     = (const float*)d_in[16];
  (void)in_sizes; (void)n_in; (void)out_size; (void)ws_size;

  float* out = (float*)d_out;
  float* c = out;                       // canvas lives in d_out[0 .. B*HW)

  // ---- workspace arena (256B aligned chunks)
  char* wsp = (char*)d_ws;
  auto arena = [&](size_t bytes) -> void* {
    void* p = (void*)wsp;
    wsp += (bytes + 255) & ~(size_t)255;
    return p;
  };
  __bf16* Wenc = (__bf16*)arena((size_t)4 * L_ * KENC_ * 2);  // (4096, 2336)
  __bf16* Wdec = (__bf16*)arena((size_t)4 * L_ * KDEC_ * 2);  // (4096, 1152)
  __bf16* Wz   = (__bf16*)arena((size_t)2 * Z_ * L_ * 2);     // (256, 1024)
  __bf16* Aenc = (__bf16*)arena((size_t)B_ * KENC_ * 2);
  __bf16* Adec = (__bf16*)arena((size_t)B_ * KDEC_ * 2);
  __bf16* Ah   = (__bf16*)arena((size_t)B_ * L_ * 2);
  float* xhat  = (float*)arena((size_t)B_ * HW_ * 4);
  float* G     = (float*)arena((size_t)B_ * 4 * L_ * 4);
  float* states = (float*)arena((size_t)4 * B_ * L_ * 4);     // henc,cenc,hdec,cdec
  float* henc = states;
  float* cenc = states + (size_t)B_ * L_;
  float* hdec = states + (size_t)2 * B_ * L_;
  float* cdec = states + (size_t)3 * B_ * L_;
  float* r     = (float*)arena((size_t)B_ * 288 * 4);
  float* stats = (float*)arena((size_t)B_ * 2 * Z_ * 4);
  float* zbuf  = (float*)arena((size_t)B_ * Z_ * 4);
  float* Fx    = (float*)arena((size_t)B_ * NF_ * 64 * 4);
  float* Fy    = (float*)arena((size_t)B_ * NF_ * 64 * 4);
  float* gam   = (float*)arena((size_t)B_ * 4);
  float* wbuf  = (float*)arena((size_t)B_ * NF_ * NF_ * 4);
  float* ra    = (float*)arena((size_t)B_ * 5 * 4);
  float* wa    = (float*)arena((size_t)B_ * 5 * 4);

  const int TPB = 256;
  auto blocks = [](size_t n, int tpb) { return (unsigned)((n + tpb - 1) / tpb); };

  const size_t shEnc = (size_t)32 * KENC_ * 2;   // dynamic LDS for A tiles
  const size_t shDec = (size_t)32 * KDEC_ * 2;
  const size_t shZ   = (size_t)32 * L_ * 2;

  // ---- one-time (per launch) weight conversion + state init
  k_cvt2<<<blocks((size_t)4 * L_ * KENC_, TPB), TPB, 0, stream>>>(encWih, 1312, encWhh, L_, Wenc, 4 * L_);
  k_cvt2<<<blocks((size_t)4 * L_ * KDEC_, TPB), TPB, 0, stream>>>(decWih, Z_, decWhh, L_, Wdec, 4 * L_);
  k_cvt<<<blocks((size_t)2 * Z_ * L_, TPB), TPB, 0, stream>>>(zW, Wz, (size_t)2 * Z_ * L_);
  k_zero<<<blocks((size_t)4 * B_ * L_, TPB), TPB, 0, stream>>>(states, (size_t)4 * B_ * L_);
  k_zero<<<blocks((size_t)B_ * HW_, TPB), TPB, 0, stream>>>(c, (size_t)B_ * HW_);

  for (int t = 0; t < T_; ++t) {
    // x_hat = x - sigmoid(c)
    k_xhat<<<blocks((size_t)B_ * HW_, TPB), TPB, 0, stream>>>(x, c, xhat);
    // read attention params + filterbank + glimpse
    k_linear<<<blocks((size_t)B_ * 5, 128), 128, 0, stream>>>(hdec, raW, rab, ra, 5, L_);
    k_filterbank<<<B_ * NF_, 64, 0, stream>>>(ra, Fx, Fy, gam, 1.0f);
    k_read<<<B_, 256, 0, stream>>>(x, xhat, Fx, Fy, gam, r);
    // encoder LSTM (fused GEMM over K = 288 + 1024 + 1024)
    k_pack_enc<<<blocks((size_t)B_ * KENC_, TPB), TPB, 0, stream>>>(r, hdec, henc, Aenc);
    k_gemm_bf16<<<dim3(B_ / 32, (4 * L_) / 256), 256, shEnc, stream>>>(Aenc, Wenc, G, B_, 4 * L_, KENC_);
    k_lstm<<<blocks((size_t)B_ * L_, TPB), TPB, 0, stream>>>(G, enc_bih, enc_bhh, henc, cenc);
    // latent sample
    k_cvt<<<blocks((size_t)B_ * L_, TPB), TPB, 0, stream>>>(henc, Ah, (size_t)B_ * L_);
    k_gemm_bf16<<<dim3(B_ / 32, (2 * Z_) / 256), 256, shZ, stream>>>(Ah, Wz, stats, B_, 2 * Z_, L_);
    k_z<<<blocks((size_t)B_ * Z_, TPB), TPB, 0, stream>>>(stats, zb, zbuf, out, t);
    // decoder LSTM (fused GEMM over K = 128 + 1024)
    k_pack_dec<<<blocks((size_t)B_ * KDEC_, TPB), TPB, 0, stream>>>(zbuf, hdec, Adec);
    k_gemm_bf16<<<dim3(B_ / 32, (4 * L_) / 256), 256, shDec, stream>>>(Adec, Wdec, G, B_, 4 * L_, KDEC_);
    k_lstm<<<blocks((size_t)B_ * L_, TPB), TPB, 0, stream>>>(G, dec_bih, dec_bhh, hdec, cdec);
    // write
    k_linear<<<blocks((size_t)B_ * NF_ * NF_, 128), 128, 0, stream>>>(hdec, wW, wb, wbuf, NF_ * NF_, L_);
    k_linear<<<blocks((size_t)B_ * 5, 128), 128, 0, stream>>>(hdec, waW, wab, wa, 5, L_);
    k_filterbank<<<B_ * NF_, 64, 0, stream>>>(wa, Fx, Fy, gam, -1.0f);
    k_write<<<B_, 256, 0, stream>>>(wbuf, Fx, Fy, gam, c);
  }
}
